// PathQualityAwareConv_40303973105672
// MI455X (gfx1250) — matrix operand order repaired and checked by reading, not compiled, not run on hardware
//
#include <hip/hip_runtime.h>
#include <hip/hip_bf16.h>

typedef __attribute__((ext_vector_type(16))) __bf16 v16bf;
typedef __attribute__((ext_vector_type(8)))  float  v8f;

union frag32 {           // one WMMA bf16 operand fragment per lane: 32 bytes
    v16bf bf;
    uint4 q[2];
};

#define N_NODES 50000
#define N_EDGES 800000
#define NEG_SLOPE 0.2f

// ---- monotonic float <-> uint key for atomicMax-based segment max ----
__device__ __forceinline__ unsigned fkey(float x) {
    unsigned b = __float_as_uint(x);
    return (b & 0x80000000u) ? ~b : (b | 0x80000000u);
}
__device__ __forceinline__ float fdec(unsigned k) {
    unsigned b = (k & 0x80000000u) ? (k & 0x7FFFFFFFu) : ~k;
    return __uint_as_float(b);
}

// ---- K0a: fold W_edge @ att_edge into [16,4]; adaptive pool 8->16 ----
__global__ void k_setup(const float* __restrict__ W_edge,
                        const float* __restrict__ att_edge,
                        const float* __restrict__ nstate,
                        float* __restrict__ w_eff, float* __restrict__ cf) {
    int t = threadIdx.x;
    if (t < 64) {
        int d = t >> 2, h = t & 3;
        float s = 0.f;
        for (int c = 0; c < 64; ++c)
            s += W_edge[d * 256 + h * 64 + c] * att_edge[h * 64 + c];
        w_eff[d * 4 + h] = s;
    } else if (t < 80) {
        int i = t - 64;
        cf[i] = nstate[i >> 1];   // adaptive_avg_pool1d(8 -> 16) == repeat
    }
}

// ---- K0b: bf16 repack: xb row-major, Wt = W_gat^T, Wct = W_c^T zero-padded ----
#define XB_ELEMS   (N_NODES * 128)        // 6,400,000
#define WT_ELEMS   (256 * 128)            // 32,768
#define WCT_ELEMS  (64 * 96)              // 6,144
__global__ void k_prep(const float* __restrict__ x,
                       const float* __restrict__ Wg,
                       const float* __restrict__ Wc,
                       __bf16* __restrict__ xb,
                       __bf16* __restrict__ Wt,
                       __bf16* __restrict__ Wct) {
    int idx = blockIdx.x * blockDim.x + threadIdx.x;
    if (idx < XB_ELEMS) {
        xb[idx] = (__bf16)x[idx];
    } else if (idx < XB_ELEMS + WT_ELEMS) {
        int t = idx - XB_ELEMS;
        int n = t >> 7, k = t & 127;                 // Wt[n][k] = Wg[k][n]
        Wt[t] = (__bf16)Wg[k * 256 + n];
    } else if (idx < XB_ELEMS + WT_ELEMS + WCT_ELEMS) {
        int t = idx - XB_ELEMS - WT_ELEMS;
        int n = t / 96, k = t - n * 96;              // Wct[n][k], k padded to 96
        Wct[t] = (__bf16)((k < 80) ? Wc[k * 64 + n] : 0.f);
    }
}

// ---- K1: xs = x @ W_gat  via bf16 WMMA; 4 C-tiles per wave, b128 fragments ----
// A lane frag: rows of xb, K runs [kb+8*half, +8) and [kb+16+8*half, +8)  (table order)
// B lane frag: Wt[ncol], K runs [kb+8*half, +8) and [kb+16+8*half, +8)  (table order)
__global__ __launch_bounds__(128) void k_xs_wmma(const __bf16* __restrict__ xb,
                                                 const __bf16* __restrict__ Wt,
                                                 float* __restrict__ xs) {
    const int lane = threadIdx.x & 31;
    const int wv   = threadIdx.x >> 5;        // 0..3 -> col tiles wv*4 .. wv*4+3
    const int rt   = blockIdx.x;              // row tile (N/16 = 3125, exact)
    const int m    = lane & 15, half = lane >> 4;
    const int row  = rt * 16 + m;
    if (rt + 1 < 3125)                        // prefetch next row tile of A
        __builtin_prefetch(xb + (row + 16) * 128, 0, 3);
    v8f acc[4] = {{}, {}, {}, {}};
    for (int kb = 0; kb < 128; kb += 32) {
        frag32 a;
        const __bf16* ar = xb + row * 128 + kb + 8 * half;
        a.q[0] = *(const uint4*)(ar);
        a.q[1] = *(const uint4*)(ar + 16);
#pragma unroll
        for (int t = 0; t < 4; ++t) {
            const int ncol = (wv * 4 + t) * 16 + m;
            frag32 b;
            const __bf16* br = Wt + ncol * 128 + kb + 8 * half;
            b.q[0] = *(const uint4*)(br);
            b.q[1] = *(const uint4*)(br + 16);
            acc[t] = __builtin_amdgcn_wmma_f32_16x16x32_bf16(
                false, a.bf, false, b.bf, (short)0, acc[t], false, false);
        }
    }
#pragma unroll
    for (int t = 0; t < 4; ++t)
#pragma unroll
        for (int r = 0; r < 8; ++r)
            xs[(rt * 16 + r + 8 * half) * 256 + (wv * 4 + t) * 16 + m] = acc[t][r];
}

// ---- K2: per-node attention scores a_src/a_dst [N,4] ----
__global__ void k_scores(const float* __restrict__ xs,
                         const float* __restrict__ att_src,
                         const float* __restrict__ att_dst,
                         float* __restrict__ a_src, float* __restrict__ a_dst) {
    int idx = blockIdx.x * blockDim.x + threadIdx.x;
    if (idx >= N_NODES * 4) return;
    int n = idx >> 2, h = idx & 3;
    const float* xr = xs + n * 256 + h * 64;
    float ss = 0.f, sd = 0.f;
    for (int c = 0; c < 64; ++c) {
        float v = xr[c];
        ss += v * att_src[h * 64 + c];
        sd += v * att_dst[h * 64 + c];
    }
    a_src[idx] = ss;
    a_dst[idx] = sd;
}

// ---- K3: edge MLP + alpha + atomic segment-max ----
__global__ void k_edge_alpha(const int* __restrict__ ei,
                             const float* __restrict__ edge_attr,
                             const float* __restrict__ quality,
                             const float* __restrict__ W_q1, const float* __restrict__ b_q1,
                             const float* __restrict__ W_q2, const float* __restrict__ b_q2,
                             const float* __restrict__ w_eff,
                             const float* __restrict__ a_src, const float* __restrict__ a_dst,
                             float* __restrict__ alpha, unsigned* __restrict__ amax) {
    int e = blockIdx.x * blockDim.x + threadIdx.x;
    if (e >= N_EDGES) return;
    int src = ei[e], dst = ei[N_EDGES + e];
    float qr[4];
#pragma unroll
    for (int i = 0; i < 4; ++i) qr[i] = quality[e * 4 + i];
    float hm[8];
#pragma unroll
    for (int j = 0; j < 8; ++j) {
        float s = b_q1[j];
#pragma unroll
        for (int i = 0; i < 4; ++i) s += qr[i] * W_q1[i * 8 + j];
        hm[j] = fmaxf(s, 0.f);
    }
    float ae[4] = {0.f, 0.f, 0.f, 0.f};
#pragma unroll
    for (int d = 0; d < 16; ++d) {
        float q = b_q2[d];
#pragma unroll
        for (int j = 0; j < 8; ++j) q += hm[j] * W_q2[j * 16 + d];
        float enh = edge_attr[e * 16 + d] + q;
#pragma unroll
        for (int h = 0; h < 4; ++h) ae[h] += enh * w_eff[d * 4 + h];
    }
#pragma unroll
    for (int h = 0; h < 4; ++h) {
        float al = a_src[src * 4 + h] + a_dst[dst * 4 + h] + ae[h];
        al = al > 0.f ? al : NEG_SLOPE * al;
        alpha[e * 4 + h] = al;
        atomicMax(&amax[dst * 4 + h], fkey(al));
    }
}

// ---- K4: ex = exp(alpha - max), denom += ex (in-place overwrite alpha) ----
__global__ void k_edge_exp(const int* __restrict__ ei,
                           float* __restrict__ alpha_ex,
                           const unsigned* __restrict__ amax,
                           float* __restrict__ denom) {
    int idx = blockIdx.x * blockDim.x + threadIdx.x;
    if (idx >= N_EDGES * 4) return;
    int e = idx >> 2, h = idx & 3;
    int dst = ei[N_EDGES + e];
    float ex = __expf(alpha_ex[idx] - fdec(amax[dst * 4 + h]));
    alpha_ex[idx] = ex;
    atomicAdd(&denom[dst * 4 + h], ex);
}

// ---- K5: weighted scatter msg[dst] += ex * xs[src]; 64 lanes per edge ----
__global__ void k_edge_scatter(const int* __restrict__ ei,
                               const float* __restrict__ ex,
                               const float* __restrict__ xs,
                               float* __restrict__ msg) {
    long long idx = (long long)blockIdx.x * blockDim.x + threadIdx.x;
    if (idx >= (long long)N_EDGES * 64) return;
    int e = (int)(idx >> 6), c = (int)(idx & 63);
    int src = ei[e], dst = ei[N_EDGES + e];
    const float* xr = xs + src * 256 + c;
    float* mr = msg + dst * 256 + c;
    atomicAdd(mr +   0, ex[e * 4 + 0] * xr[  0]);
    atomicAdd(mr +  64, ex[e * 4 + 1] * xr[ 64]);
    atomicAdd(mr + 128, ex[e * 4 + 2] * xr[128]);
    atomicAdd(mr + 192, ex[e * 4 + 3] * xr[192]);
}

// ---- K6: normalize + head-mean + concat(cf) + @W_c + ReLU, WMMA ----
// K padded 80 -> 96 (3 bf16 WMMA steps). One wave per 16x16 out tile.
__global__ __launch_bounds__(128) void k_final_wmma(const float* __restrict__ msg,
                                                    const float* __restrict__ denom,
                                                    const float* __restrict__ cf,
                                                    const float* __restrict__ b_gat,
                                                    const __bf16* __restrict__ Wct,
                                                    const float* __restrict__ b_c,
                                                    float* __restrict__ out) {
    const int lane = threadIdx.x & 31;
    const int jt   = threadIdx.x >> 5;   // col tile 0..3
    const int rt   = blockIdx.x;         // row tile 0..3124
    const int m    = lane & 15, half = lane >> 4;
    const int row  = rt * 16 + m;
    const int ncol = jt * 16 + m;
    float rd[4];
#pragma unroll
    for (int h = 0; h < 4; ++h)
        rd[h] = 0.25f / (denom[row * 4 + h] + 1e-16f);   // head-mean folded in
    v8f acc = {};
    for (int kb = 0; kb < 96; kb += 32) {
        frag32 a, b;
#pragma unroll
        for (int j = 0; j < 16; ++j) {    // A values are computed, not loaded raw
            int p = j >> 1, lo = j & 1;
            int ka = kb + (p < 4 ? 8 * half + 2 * p + lo
                                 : 16 + 8 * half + 2 * (p - 4) + lo);
            float av;
            if (ka < 64) {
                const float* mr = msg + row * 256 + ka;
                av = mr[0] * rd[0] + mr[64] * rd[1] + mr[128] * rd[2] +
                     mr[192] * rd[3] + b_gat[ka];
            } else if (ka < 80) {
                av = cf[ka - 64];
            } else {
                av = 0.f;
            }
            a.bf[j] = (__bf16)av;
        }
        const __bf16* br = Wct + ncol * 96 + kb + 8 * half;
        b.q[0] = *(const uint4*)(br);
        b.q[1] = *(const uint4*)(br + 16);
        acc = __builtin_amdgcn_wmma_f32_16x16x32_bf16(
            false, a.bf, false, b.bf, (short)0, acc, false, false);
    }
#pragma unroll
    for (int r = 0; r < 8; ++r) {
        int rr = rt * 16 + r + 8 * half;
        int col = jt * 16 + m;
        out[rr * 64 + col] = fmaxf(acc[r] + b_c[col], 0.f);
    }
}

extern "C" void kernel_launch(void* const* d_in, const int* in_sizes, int n_in,
                              void* d_out, int out_size, void* d_ws, size_t ws_size,
                              hipStream_t stream) {
    (void)in_sizes; (void)n_in; (void)out_size; (void)ws_size;
    const float* x        = (const float*)d_in[0];
    const int*   ei       = (const int*)  d_in[1];
    const float* edge_atr = (const float*)d_in[2];
    const float* quality  = (const float*)d_in[3];
    const float* nstate   = (const float*)d_in[4];
    const float* W_gat    = (const float*)d_in[5];
    const float* att_src  = (const float*)d_in[6];
    const float* att_dst  = (const float*)d_in[7];
    const float* W_edge   = (const float*)d_in[8];
    const float* att_edge = (const float*)d_in[9];
    const float* b_gat    = (const float*)d_in[10];
    const float* W_q1     = (const float*)d_in[11];
    const float* b_q1     = (const float*)d_in[12];
    const float* W_q2     = (const float*)d_in[13];
    const float* b_q2     = (const float*)d_in[14];
    const float* W_c      = (const float*)d_in[15];
    const float* b_c      = (const float*)d_in[16];

    // workspace layout (floats): ~131 MB total
    float*    ws    = (float*)d_ws;
    float*    xs    = ws;                          // 50000*256 f32
    float*    msg   = xs    + 12800000;            // 50000*256 f32
    float*    alpha = msg   + 12800000;            // 800000*4 f32 (reused as ex)
    float*    a_src = alpha + 3200000;             // 50000*4
    float*    a_dst = a_src + 200000;              // 50000*4
    float*    denom = a_dst + 200000;              // 50000*4
    unsigned* amax  = (unsigned*)(denom + 200000); // 50000*4 uint keys
    float*    w_eff = (float*)(amax + 200000);     // 16*4
    float*    cf    = w_eff + 64;                  // 16 (+pad to 29600080 floats)
    __bf16*   xb    = (__bf16*)(ws + 29600080);    // 6,400,000 bf16
    __bf16*   Wt    = xb + XB_ELEMS;               // 32,768 bf16
    __bf16*   Wct   = Wt + WT_ELEMS;               // 6,144 bf16

    // zero accumulators every call (graph-replay safe): msg, denom, amax(key 0 = -inf)
    hipMemsetAsync(msg,   0, 12800000 * sizeof(float), stream);
    hipMemsetAsync(denom, 0,   400000 * sizeof(float), stream); // denom + amax contiguous

    k_setup<<<1, 128, 0, stream>>>(W_edge, att_edge, nstate, w_eff, cf);
    k_prep<<<(XB_ELEMS + WT_ELEMS + WCT_ELEMS + 255) / 256, 256, 0, stream>>>(
        x, W_gat, W_c, xb, Wt, Wct);
    k_xs_wmma<<<3125, 128, 0, stream>>>(xb, Wt, xs);
    k_scores<<<(N_NODES * 4 + 255) / 256, 256, 0, stream>>>(xs, att_src, att_dst,
                                                            a_src, a_dst);
    k_edge_alpha<<<(N_EDGES + 255) / 256, 256, 0, stream>>>(
        ei, edge_atr, quality, W_q1, b_q1, W_q2, b_q2, w_eff, a_src, a_dst,
        alpha, amax);
    k_edge_exp<<<(N_EDGES * 4 + 255) / 256, 256, 0, stream>>>(ei, alpha, amax, denom);
    k_edge_scatter<<<(N_EDGES * 64 + 255) / 256, 256, 0, stream>>>(ei, alpha, xs, msg);
    k_final_wmma<<<3125, 128, 0, stream>>>(msg, denom, cf, b_gat, Wct, b_c,
                                           (float*)d_out);
}
